// Guide_57286273794829
// MI455X (gfx1250) — compile-verified, matrix operand order checked by reading, not run on hardware
//
#include <hip/hip_runtime.h>
#include <hip/hip_bf16.h>
#include <math.h>

// ---- problem constants (from reference) ----
#define BS    64
#define KK_N  4
#define PTS   5
#define RES   28
#define STEPS 500
#define EPSV  1e-06f

#define TDIM   512          // padded K (t) dimension for WMMA (500 -> 512)
#define STRIDE 520          // LDS row stride in halves: 260 dwords == 4 mod 64
                            //  -> conflict-free ds_load_b128 fragment loads
#define NROW   29           // 28 valid coord rows + 1 zero row
#define NPIX   (RES*RES)    // 784

typedef __attribute__((ext_vector_type(16))) _Float16 v16h;
typedef __attribute__((ext_vector_type(8)))  _Float16 v8h;
typedef __attribute__((ext_vector_type(8)))  float    v8f;

__device__ __forceinline__ unsigned pack_f16x2(float a, float b) {
    _Float16 ha = (_Float16)a, hb = (_Float16)b;
    unsigned short ua = __builtin_bit_cast(unsigned short, ha);
    unsigned short ub = __builtin_bit_cast(unsigned short, hb);
    return (unsigned)ua | ((unsigned)ub << 16);
}

// One stroke (b,kk): 256 threads = 8 wave32.
// Waves: tile = wv&3 (my=bit1, nx=bit0), khalf = wv>>2 (k-range half).
// khalf==0 waves end holding the full normalized 16x16 tile fragment.
__device__ __forceinline__ void stroke_tile(
    int b, int kk,
    const float* zp_ptr, const float* zw_ptr, const float* zwh_ptr,
    float inv2s2, float slope_strk,
    _Float16* Ey, _Float16* Ex, float* partsum, float* redmax,
    float out_vals[8])
{
    const int tid   = threadIdx.x;
    const int lane  = tid & 31;
    const int wv    = tid >> 5;          // 0..7
    const int tile  = wv & 3;
    const int khalf = wv >> 2;           // 0 or 1
    const int my    = (tile >> 1) & 1;
    const int nx    = tile & 1;
    const int H     = lane >> 4;
    const int m     = lane & 15;

    const int bk = b * KK_N + kk;

    // ---- per-thread stroke params ----
    const float s   = zwh_ptr[bk * 3 + 0];
    const float shx = zwh_ptr[bk * 3 + 1];
    const float shy = zwh_ptr[bk * 3 + 2];
    float px[PTS], py[PTS];
#pragma unroll
    for (int p = 0; p < PTS; ++p) {
        px[p] = zw_ptr[(bk * PTS + p) * 2 + 0] * s + shx;
        py[p] = zw_ptr[(bk * PTS + p) * 2 + 1] * s + shy;
    }

    // ---- phase 1: fill Ey/Ex (f16, layout [coord][t], packed pair stores) ----
    // 256 threads x 1 t-pair each covers t = 0..511.
    {
        const int t0 = tid * 2;          // even t
        float cx0 = 0.f, cy0 = 0.f, cx1 = 0.f, cy1 = 0.f;
#pragma unroll
        for (int j = 0; j < 2; ++j) {
            const int t = t0 + j;
            if (t < STEPS) {
                const float u  = (float)t * (1.0f / (float)(STEPS - 1));
                const float v  = 1.0f - u;
                const float u2 = u * u, v2 = v * v;
                const float w0 = v2 * v2;
                const float w1 = 4.0f * u * v2 * v;
                const float w2 = 6.0f * u2 * v2;
                const float w3 = 4.0f * u2 * u * v;
                const float w4 = u2 * u2;
                const float cx = w0*px[0]+w1*px[1]+w2*px[2]+w3*px[3]+w4*px[4];
                const float cy = w0*py[0]+w1*py[1]+w2*py[2]+w3*py[3]+w4*py[4];
                if (j == 0) { cx0 = cx; cy0 = cy; } else { cx1 = cx; cy1 = cy; }
            }
        }
        const bool v0 = (t0 < STEPS), v1 = (t0 + 1 < STEPS);
#pragma unroll 4
        for (int c = 0; c < RES; ++c) {
            const float g = (float)c * (1.0f / (float)(RES - 1));
            float ex0 = 0.f, ex1 = 0.f, ey0 = 0.f, ey1 = 0.f;
            if (v0) {
                const float dx = g - cx0, dy = g - cy0;
                ex0 = __expf(-dx * dx * inv2s2);
                ey0 = __expf(-dy * dy * inv2s2);
            }
            if (v1) {
                const float dx = g - cx1, dy = g - cy1;
                ex1 = __expf(-dx * dx * inv2s2);
                ey1 = __expf(-dy * dy * inv2s2);
            }
            *(unsigned*)(Ex + c * STRIDE + t0) = pack_f16x2(ex0, ex1);
            *(unsigned*)(Ey + c * STRIDE + t0) = pack_f16x2(ey0, ey1);
        }
        // zero row (row 28), read by out-of-range tile rows 28..31
        *(unsigned*)(Ex + 28 * STRIDE + t0) = 0u;
        *(unsigned*)(Ey + 28 * STRIDE + t0) = 0u;
    }
    __syncthreads();

    // ---- phase 2: WMMA  D(32x32) = Ey(32x512) * Ex^T(512x32), k split /2 ----
    int ra = my * 16 + m; if (ra > 28) ra = 28;   // clamp to zero row
    int rb = nx * 16 + m; if (rb > 28) rb = 28;
    const _Float16* pa = Ey + ra * STRIDE;
    const _Float16* pb = Ex + rb * STRIDE;

    v8f acc = {};
#pragma unroll
    for (int ks = 0; ks < 8; ++ks) {
        const int kb = (khalf * 8 + ks) * 32 + H * 8;
        v8h alo = *(const v8h*)(pa + kb);
        v8h ahi = *(const v8h*)(pa + kb + 16);
        v8h blo = *(const v8h*)(pb + kb);
        v8h bhi = *(const v8h*)(pb + kb + 16);
        v16h a, bm;
#pragma unroll
        for (int i = 0; i < 8; ++i) {
            a[i] = alo[i];  a[i + 8] = ahi[i];
            bm[i] = blo[i]; bm[i + 8] = bhi[i];
        }
        acc = __builtin_amdgcn_wmma_f32_16x16x32_f16(
            false, a, false, bm, (short)0, acc, false, false);
    }

    // khalf==1 waves publish partial accumulators
    if (khalf == 1) {
#pragma unroll
        for (int i = 0; i < 8; ++i)
            partsum[(tile * 32 + lane) * 8 + i] = acc[i];
    }
    __syncthreads();

    // ---- phase 3 (khalf==0 waves): combine, z_pres, max-norm, tanh ----
    if (khalf == 0) {
#pragma unroll
        for (int i = 0; i < 8; ++i)
            acc[i] += partsum[(tile * 32 + lane) * 8 + i];

        float lm = 0.0f;
#pragma unroll
        for (int i = 0; i < 8; ++i) lm = fmaxf(lm, acc[i]);
#pragma unroll
        for (int off = 16; off >= 1; off >>= 1)
            lm = fmaxf(lm, __shfl_xor(lm, off, 32));
        if (lane == 0) redmax[tile] = lm;
    }
    __syncthreads();

    if (khalf == 0) {
        const float zp = zp_ptr[bk];
        const float wm = fmaxf(fmaxf(redmax[0], redmax[1]),
                               fmaxf(redmax[2], redmax[3])) * zp;
        const float itan = 1.0f / tanhf(slope_strk);
        const float rmax = 1.0f / (wm + EPSV);
#pragma unroll
        for (int i = 0; i < 8; ++i)
            out_vals[i] = tanhf(acc[i] * zp * rmax * slope_strk) * itan;
    }
}

// ---- split path: one WG per (b,k) stroke -> workspace ----
__global__ void __launch_bounds__(256)
stroke_kernel(const float* __restrict__ z_pres,
              const float* __restrict__ z_what,
              const float* __restrict__ z_where,
              const float* __restrict__ sigma_p,
              const float* __restrict__ slope_strk_p,
              float* __restrict__ ws)
{
    __shared__ _Float16 Ey[NROW * STRIDE];
    __shared__ _Float16 Ex[NROW * STRIDE];
    __shared__ float partsum[4 * 32 * 8];
    __shared__ float redmax[4];

    const int bk = blockIdx.x;
    const int b = bk >> 2, kk = bk & 3;
    const float sg = sigma_p[0];
    const float inv2s2 = 1.0f / (2.0f * sg * sg);

    float ov[8] = {0,0,0,0,0,0,0,0};
    stroke_tile(b, kk, z_pres, z_what, z_where, inv2s2, slope_strk_p[0],
                Ey, Ex, partsum, redmax, ov);

    const int lane = threadIdx.x & 31;
    const int wv   = threadIdx.x >> 5;
    if ((wv >> 2) == 0) {
        const int my = (wv >> 1) & 1, nx = wv & 1;
        const int H = lane >> 4, m = lane & 15;
        const int x = nx * 16 + m;
#pragma unroll
        for (int i = 0; i < 8; ++i) {
            const int y = my * 16 + H * 8 + i;
            if (y < RES && x < RES)
                ws[bk * NPIX + y * RES + x] = ov[i];
        }
    }
}

__global__ void combine_kernel(const float* __restrict__ ws,
                               const float* __restrict__ slope_p,
                               float* __restrict__ out, int n)
{
    const int idx = blockIdx.x * blockDim.x + threadIdx.x;
    if (idx >= n) return;
    const int b = idx / NPIX, p = idx % NPIX;
    float s = 0.f;
#pragma unroll
    for (int k = 0; k < KK_N; ++k)
        s += ws[(b * KK_N + k) * NPIX + p];
    const float slope = slope_p[0];
    out[idx] = tanhf(s * slope) / tanhf(slope);
}

// ---- fused fallback (no workspace): one WG per batch element ----
__global__ void __launch_bounds__(256)
fused_kernel(const float* __restrict__ z_pres,
             const float* __restrict__ z_what,
             const float* __restrict__ z_where,
             const float* __restrict__ sigma_p,
             const float* __restrict__ slope_strk_p,
             const float* __restrict__ slope_p,
             float* __restrict__ out)
{
    __shared__ _Float16 Ey[NROW * STRIDE];
    __shared__ _Float16 Ex[NROW * STRIDE];
    __shared__ float partsum[4 * 32 * 8];
    __shared__ float redmax[4];

    const int b = blockIdx.x;
    const float sg = sigma_p[0];
    const float inv2s2 = 1.0f / (2.0f * sg * sg);
    const float sstrk = slope_strk_p[0];

    float img[8] = {0.f, 0.f, 0.f, 0.f, 0.f, 0.f, 0.f, 0.f};
    for (int kk = 0; kk < KK_N; ++kk) {
        float ov[8] = {0,0,0,0,0,0,0,0};
        stroke_tile(b, kk, z_pres, z_what, z_where, inv2s2, sstrk,
                    Ey, Ex, partsum, redmax, ov);
#pragma unroll
        for (int i = 0; i < 8; ++i) img[i] += ov[i];
        __syncthreads();
    }

    const float slope = slope_p[0];
    const float itan = 1.0f / tanhf(slope);
    const int lane = threadIdx.x & 31;
    const int wv   = threadIdx.x >> 5;
    if ((wv >> 2) == 0) {
        const int my = (wv >> 1) & 1, nx = wv & 1;
        const int H = lane >> 4, m = lane & 15;
        const int x = nx * 16 + m;
#pragma unroll
        for (int i = 0; i < 8; ++i) {
            const int y = my * 16 + H * 8 + i;
            if (y < RES && x < RES)
                out[b * NPIX + y * RES + x] = tanhf(img[i] * slope) * itan;
        }
    }
}

extern "C" void kernel_launch(void* const* d_in, const int* in_sizes, int n_in,
                              void* d_out, int out_size, void* d_ws, size_t ws_size,
                              hipStream_t stream)
{
    const float* z_pres  = (const float*)d_in[0];   // (64,4)
    const float* z_what  = (const float*)d_in[1];   // (64,4,5,2)
    const float* z_where = (const float*)d_in[2];   // (64,4,3)
    const float* sigma   = (const float*)d_in[3];   // scalar
    const float* sstrk   = (const float*)d_in[4];   // scalar
    const float* slope   = (const float*)d_in[5];   // scalar
    float* out = (float*)d_out;                     // (64,1,28,28)

    const size_t need_ws = (size_t)BS * KK_N * NPIX * sizeof(float);
    if (ws_size >= need_ws) {
        float* ws = (float*)d_ws;
        stroke_kernel<<<BS * KK_N, 256, 0, stream>>>(
            z_pres, z_what, z_where, sigma, sstrk, ws);
        const int n = BS * NPIX;
        combine_kernel<<<(n + 255) / 256, 256, 0, stream>>>(ws, slope, out, n);
    } else {
        fused_kernel<<<BS, 256, 0, stream>>>(
            z_pres, z_what, z_where, sigma, sstrk, slope, out);
    }
}
